// Decoder_48404281426026
// MI455X (gfx1250) — matrix operand order repaired and checked by reading, not compiled, not run on hardware
//
#include <hip/hip_runtime.h>
#include <hip/hip_bf16.h>

// ---------------------------------------------------------------- constants
constexpr int kB  = 128;
constexpr int kT  = 256;
constexpr int kE  = 384;
constexpr int kH  = 6;
constexpr int kD  = 64;     // kE / kH
constexpr int kF  = 1536;   // DFF
constexpr int kL  = 6;
constexpr int kV  = 65;
constexpr int kBT = kB * kT;        // 32768 rows
constexpr int kVpad = 128;          // padded rows for transposed Wl

typedef __bf16 bf16_t;
typedef __attribute__((ext_vector_type(16))) __bf16 v16bf;
typedef __attribute__((ext_vector_type(8)))  __bf16 v8bf;
typedef __attribute__((ext_vector_type(8)))  float  v8f;

// ---------------------------------------------------------------- wmma core
__device__ __forceinline__ v8f wmma_bf16(v16bf a, v16bf b, v8f c) {
  // D = A(16x32 bf16) * B(32x16 bf16) + C(16x16 f32)
  return __builtin_amdgcn_wmma_f32_16x16x32_bf16(
      /*neg_a=*/false, a, /*neg_b=*/false, b, /*c_mod=*/(short)0, c,
      /*reuse_a=*/false, /*reuse_b=*/false);
}

// A-fragment (16x32 bf16): lane m = lane&15, kb = (lane<16)?0:8.
// element j<8 -> k = kb + j ; element j>=8 -> k = kb + 16 + (j-8)
__device__ __forceinline__ v16bf frag_a(const bf16_t* p) {
  v8bf lo = *(const v8bf*)(p);
  v8bf hi = *(const v8bf*)(p + 16);
  v16bf r;
#pragma unroll
  for (int j = 0; j < 8; ++j) { r[j] = lo[j]; r[j + 8] = hi[j]; }
  return r;
}
// B-fragment (32x16 bf16): lane n = lane&15, kb = (lane<16)?0:8.
// element j<8 -> k = kb + j ; element j>=8 -> k = kb + 16 + (j-8)
__device__ __forceinline__ v16bf frag_b(const bf16_t* p) {
  v8bf lo = *(const v8bf*)(p);
  v8bf hi = *(const v8bf*)(p + 16);
  v16bf r;
#pragma unroll
  for (int j = 0; j < 8; ++j) { r[j] = lo[j]; r[j + 8] = hi[j]; }
  return r;
}

// ---------------------------------------------------------------- embed
__global__ void embed_kernel(const int* __restrict__ tokens,
                             const float* __restrict__ pos,
                             const float* __restrict__ emb,
                             float* __restrict__ x) {
  size_t idx = (size_t)blockIdx.x * blockDim.x + threadIdx.x;
  if (idx >= (size_t)kBT * kE) return;
  int e = (int)(idx % kE);
  size_t bt = idx / kE;
  int t = (int)(bt % kT);
  int tok = tokens[bt];
  x[idx] = emb[(size_t)tok * kE + e] + pos[(size_t)t * kE + e];
}

// ---------------------------------------------------------------- f32 -> bf16
__global__ void f2b_kernel(const float* __restrict__ in,
                           bf16_t* __restrict__ out, size_t n) {
  size_t idx = (size_t)blockIdx.x * blockDim.x + threadIdx.x;
  if (idx < n) out[idx] = (bf16_t)in[idx];
}

// ---------------------------------------------------------------- weight transpose
// src fp32 [K, N] -> dst bf16 [Npad, K], rows >= N zero-filled
__global__ void wtrans_kernel(const float* __restrict__ src,
                              bf16_t* __restrict__ dst,
                              int K, int N, int Npad) {
  size_t idx = (size_t)blockIdx.x * blockDim.x + threadIdx.x;
  if (idx >= (size_t)Npad * K) return;
  int k = (int)(idx % K);
  int n = (int)(idx / K);
  dst[idx] = (n < N) ? (bf16_t)src[(size_t)k * N + n] : (bf16_t)0.0f;
}

// ---------------------------------------------------------------- layernorm
// one wave (32 lanes) per row of E=384
__global__ __launch_bounds__(256) void ln_kernel(const float* __restrict__ x,
                                                 const float* __restrict__ g,
                                                 const float* __restrict__ be,
                                                 bf16_t* __restrict__ out,
                                                 int rows) {
  int wid  = (int)((blockIdx.x * 256 + threadIdx.x) >> 5);
  int lane = (int)(threadIdx.x & 31);
  if (wid >= rows) return;
  const float* xr = x + (size_t)wid * kE;
  float v[12];
  float s = 0.f, s2 = 0.f;
#pragma unroll
  for (int j = 0; j < 12; ++j) {
    v[j] = xr[lane + 32 * j];
    s += v[j];
    s2 += v[j] * v[j];
  }
#pragma unroll
  for (int off = 16; off > 0; off >>= 1) {
    s  += __shfl_xor(s,  off, 32);
    s2 += __shfl_xor(s2, off, 32);
  }
  const float invE = 1.0f / (float)kE;
  float mu  = s * invE;
  float var = s2 * invE - mu * mu;
  float inv = rsqrtf(var + 1e-5f);
  bf16_t* o = out + (size_t)wid * kE;
#pragma unroll
  for (int j = 0; j < 12; ++j) {
    int c = lane + 32 * j;
    o[c] = (bf16_t)((v[j] - mu) * inv * g[c] + be[c]);
  }
}

// ---------------------------------------------------------------- GEMM
// C[M,N] = act(A_bf16[M,K] @ Wt_bf16[N,K]^T + bias) (+ resid)
// block = 256 threads = 8 waves. Block tile 128x64.
// Wave (w): m-quad = w&3 -> 32 rows, n-half = w>>2 -> 32 cols.
// 2x2 register blocking: 2 A-frags x 2 B-frags -> 4 WMMA / K-step
// (2 b128 loads per WMMA instead of 3).
__global__ __launch_bounds__(256) void gemm_kernel(
    const bf16_t* __restrict__ A,    // [M,K]
    const bf16_t* __restrict__ Wt,   // [Npad,K], Npad >= gridDim.y*64
    const float* __restrict__ bias,  // [N]
    const float* __restrict__ resid, // [M,N] or null
    float* __restrict__ outF,        // [M,N] or null
    bf16_t* __restrict__ outB,       // [M,N] or null
    int N, int K, int relu) {
  int tid   = threadIdx.x;
  int w     = tid >> 5;
  int lane  = tid & 31;
  int laneM = lane & 15;
  int hiH   = lane >> 4;

  int m0 = blockIdx.x * 128 + (w & 3) * 32;   // 32 rows per wave
  int n0 = blockIdx.y * 64 + (w >> 2) * 32;   // 32 cols per wave

  const bf16_t* a0row = A  + (size_t)(m0 + laneM) * K;
  const bf16_t* a1row = A  + (size_t)(m0 + 16 + laneM) * K;
  const bf16_t* b0row = Wt + (size_t)(n0 + laneM) * K;
  const bf16_t* b1row = Wt + (size_t)(n0 + 16 + laneM) * K;
  int a_kb = hiH ? 8 : 0;
  int b_kb = hiH ? 8 : 0;

  v8f acc00 = {};
  v8f acc01 = {};
  v8f acc10 = {};
  v8f acc11 = {};
  for (int k0 = 0; k0 < K; k0 += 32) {
    v16bf af0 = frag_a(a0row + k0 + a_kb);
    v16bf af1 = frag_a(a1row + k0 + a_kb);
    v16bf bf0 = frag_b(b0row + k0 + b_kb);
    v16bf bf1 = frag_b(b1row + k0 + b_kb);
    acc00 = wmma_bf16(af0, bf0, acc00);
    acc01 = wmma_bf16(af0, bf1, acc01);
    acc10 = wmma_bf16(af1, bf0, acc10);
    acc11 = wmma_bf16(af1, bf1, acc11);
  }

  int nI0 = n0 + laneM;
  int nI1 = n0 + 16 + laneM;
  float bv0 = (nI0 < N) ? bias[nI0] : 0.f;
  float bv1 = (nI1 < N) ? bias[nI1] : 0.f;

#pragma unroll
  for (int sub = 0; sub < 2; ++sub) {
    int mb = m0 + sub * 16 + hiH * 8;
    v8f accA = sub ? acc10 : acc00;
    v8f accB = sub ? acc11 : acc01;
#pragma unroll
    for (int r = 0; r < 8; ++r) {
      int m = mb + r;
      size_t rowOff = (size_t)m * N;
      float v0 = accA[r] + bv0;
      float v1 = accB[r] + bv1;
      if (relu) { v0 = fmaxf(v0, 0.f); v1 = fmaxf(v1, 0.f); }
      if (resid) {
        if (nI0 < N) v0 += resid[rowOff + nI0];
        if (nI1 < N) v1 += resid[rowOff + nI1];
      }
      if (outF) {
        if (nI0 < N) outF[rowOff + nI0] = v0;
        if (nI1 < N) outF[rowOff + nI1] = v1;
      }
      if (outB) {
        if (nI0 < N) outB[rowOff + nI0] = (bf16_t)v0;
        if (nI1 < N) outB[rowOff + nI1] = (bf16_t)v1;
      }
    }
  }
}

// ---------------------------------------------------------------- fused attention
// grid.x = B*H, grid.y = T/64. block = 256 threads.
// scores(64x256) = q @ k^T * scale (causal) ; softmax ; o(64x64) = P @ v
__global__ __launch_bounds__(256) void attn_kernel(
    const bf16_t* __restrict__ Qb,  // [BT, E]
    const bf16_t* __restrict__ Kb,  // [BT, E]
    const bf16_t* __restrict__ Vb,  // [BT, E]
    bf16_t* __restrict__ Ob) {      // [BT, E]
  __shared__ __align__(16) float  sS[64 * 256];   // 64 KB scores
  __shared__ __align__(16) bf16_t sP[64 * 256];   // 32 KB probs bf16
  __shared__ __align__(16) bf16_t sVt[64 * 256];  // 32 KB v^T (d-major)

  int bh = blockIdx.x;
  int b  = bh / kH;
  int h  = bh % kH;
  int rb = blockIdx.y;        // query row block (64 rows)

  int tid   = threadIdx.x;
  int w     = tid >> 5;
  int lane  = tid & 31;
  int laneM = lane & 15;
  int hiH   = lane >> 4;
  int a_kb  = hiH ? 8 : 0;
  int b_kb  = hiH ? 8 : 0;

  // ---- stage V transposed: sVt[d][t] = v[t][d]
  {
    const bf16_t* vrow = Vb + ((size_t)(b * kT + tid) * kE + h * kD);
#pragma unroll 4
    for (int d = 0; d < kD; ++d) sVt[d * 256 + tid] = vrow[d];
  }

  // ---- scores: wave w -> mrow = w>>1 (16 rows), nhalf = w&1 (128 cols)
  int mrow  = w >> 1;
  int nhalf = w & 1;
  int m0    = mrow * 16;
  const bf16_t* qrow =
      Qb + ((size_t)(b * kT + rb * 64 + m0 + laneM) * kE + h * kD);

  v8f acc[8];
#pragma unroll
  for (int i = 0; i < 8; ++i) acc[i] = (v8f){};

#pragma unroll
  for (int k0 = 0; k0 < kD; k0 += 32) {
    v16bf af = frag_a(qrow + k0 + a_kb);
#pragma unroll
    for (int i = 0; i < 8; ++i) {
      int n0 = nhalf * 128 + i * 16;
      const bf16_t* krow =
          Kb + ((size_t)(b * kT + n0 + laneM) * kE + h * kD);
      v16bf bfv = frag_b(krow + k0 + b_kb);
      acc[i] = wmma_bf16(af, bfv, acc[i]);
    }
  }

  const float scale = 0.125f;  // 1/sqrt(64)
#pragma unroll
  for (int i = 0; i < 8; ++i) {
    int n = nhalf * 128 + i * 16 + laneM;
#pragma unroll
    for (int r = 0; r < 8; ++r) {
      int m  = m0 + hiH * 8 + r;
      int tq = rb * 64 + m;
      sS[m * 256 + n] = (n <= tq) ? acc[i][r] * scale : -3.0e38f;
    }
  }
  __syncthreads();

  // ---- softmax: thread t<64 owns row t
  if (tid < 64) {
    float mx = -3.0e38f;
    for (int j = 0; j < 256; ++j) mx = fmaxf(mx, sS[tid * 256 + j]);
    float sum = 0.f;
    for (int j = 0; j < 256; ++j) {
      float e = __expf(sS[tid * 256 + j] - mx);
      sS[tid * 256 + j] = e;
      sum += e;
    }
    float inv = 1.0f / sum;
    for (int j = 0; j < 256; ++j)
      sP[tid * 256 + j] = (bf16_t)(sS[tid * 256 + j] * inv);
  }
  __syncthreads();

  // ---- o = P(64x256) @ v(256x64): wave -> mrow2 = w&3, npair2 = w>>2
  int mrow2  = w & 3;
  int npair2 = w >> 2;
  int m02 = mrow2 * 16;
  int n02 = npair2 * 32;
  const bf16_t* arow2 = sP  + (size_t)(m02 + laneM) * 256;
  const bf16_t* b0row = sVt + (size_t)(n02 + laneM) * 256;
  const bf16_t* b1row = sVt + (size_t)(n02 + 16 + laneM) * 256;

  v8f oa0 = {};
  v8f oa1 = {};
#pragma unroll
  for (int k0 = 0; k0 < 256; k0 += 32) {
    v16bf af = frag_a(arow2 + k0 + a_kb);
    v16bf b0 = frag_b(b0row + k0 + b_kb);
    v16bf b1 = frag_b(b1row + k0 + b_kb);
    oa0 = wmma_bf16(af, b0, oa0);
    oa1 = wmma_bf16(af, b1, oa1);
  }

  int nA = n02 + laneM;
  int nB = n02 + 16 + laneM;
#pragma unroll
  for (int r = 0; r < 8; ++r) {
    int m = m02 + hiH * 8 + r;
    size_t row = (size_t)(b * kT + rb * 64 + m) * kE + h * kD;
    Ob[row + nA] = (bf16_t)oa0[r];
    Ob[row + nB] = (bf16_t)oa1[r];
  }
}

// ---------------------------------------------------------------- host side
extern "C" void kernel_launch(void* const* d_in, const int* in_sizes, int n_in,
                              void* d_out, int out_size, void* d_ws,
                              size_t ws_size, hipStream_t stream) {
  const int*   tokens = (const int*)d_in[0];
  const float* pos    = (const float*)d_in[1];
  const float* emb    = (const float*)d_in[2];
  const float* Wq = (const float*)d_in[3];
  const float* bq = (const float*)d_in[4];
  const float* Wk = (const float*)d_in[5];
  const float* bk = (const float*)d_in[6];
  const float* Wv = (const float*)d_in[7];
  const float* bv = (const float*)d_in[8];
  const float* Wo = (const float*)d_in[9];
  const float* bo = (const float*)d_in[10];
  const float* g1 = (const float*)d_in[11];
  const float* be1 = (const float*)d_in[12];
  const float* W1 = (const float*)d_in[13];
  const float* c1 = (const float*)d_in[14];
  const float* W2 = (const float*)d_in[15];
  const float* c2 = (const float*)d_in[16];
  const float* g2 = (const float*)d_in[17];
  const float* be2 = (const float*)d_in[18];
  const float* Wl = (const float*)d_in[19];
  const float* bl = (const float*)d_in[20];
  float* out = (float*)d_out;

  char* ws = (char*)d_ws;
  size_t off = 0;
  auto alloc = [&](size_t bytes) -> void* {
    void* p = ws + off;
    off += (bytes + 255) & ~(size_t)255;
    return p;
  };

  float*  x    = (float*)alloc((size_t)kBT * kE * 4);
  bf16_t* hbf  = (bf16_t*)alloc((size_t)kBT * kE * 2);
  bf16_t* qb   = (bf16_t*)alloc((size_t)kBT * kE * 2);
  bf16_t* kb   = (bf16_t*)alloc((size_t)kBT * kE * 2);
  bf16_t* vb   = (bf16_t*)alloc((size_t)kBT * kE * 2);
  bf16_t* ob   = (bf16_t*)alloc((size_t)kBT * kE * 2);
  bf16_t* midb = (bf16_t*)alloc((size_t)kBT * kF * 2);
  bf16_t* Wqt  = (bf16_t*)alloc((size_t)kL * kE * kE * 2);
  bf16_t* Wkt  = (bf16_t*)alloc((size_t)kL * kE * kE * 2);
  bf16_t* Wvt  = (bf16_t*)alloc((size_t)kL * kE * kE * 2);
  bf16_t* Wot  = (bf16_t*)alloc((size_t)kL * kE * kE * 2);
  bf16_t* W1t  = (bf16_t*)alloc((size_t)kL * kE * kF * 2);
  bf16_t* W2t  = (bf16_t*)alloc((size_t)kL * kE * kF * 2);
  bf16_t* Wlt  = (bf16_t*)alloc((size_t)kVpad * kE * 2);

  auto wtrans = [&](const float* src, bf16_t* dst, int K, int N, int Npad) {
    size_t tot = (size_t)Npad * K;
    int blocks = (int)((tot + 255) / 256);
    wtrans_kernel<<<blocks, 256, 0, stream>>>(src, dst, K, N, Npad);
  };
  for (int l = 0; l < kL; ++l) {
    size_t oEE = (size_t)l * kE * kE;
    size_t oEF = (size_t)l * kE * kF;
    wtrans(Wq + oEE, Wqt + oEE, kE, kE, kE);
    wtrans(Wk + oEE, Wkt + oEE, kE, kE, kE);
    wtrans(Wv + oEE, Wvt + oEE, kE, kE, kE);
    wtrans(Wo + oEE, Wot + oEE, kE, kE, kE);
    wtrans(W1 + oEF, W1t + oEF, kE, kF, kF);   // [E,F] -> [F,E]
    wtrans(W2 + oEF, W2t + oEF, kF, kE, kE);   // [F,E] -> [E,F]
  }
  wtrans(Wl, Wlt, kE, kV, kVpad);              // [E,V] -> [Vpad,E]

  {
    size_t tot = (size_t)kBT * kE;
    embed_kernel<<<(int)((tot + 255) / 256), 256, 0, stream>>>(tokens, pos,
                                                               emb, x);
  }

  auto gemm = [&](const bf16_t* A, const bf16_t* Wt, const float* bias,
                  const float* resid, float* oF, bf16_t* oB, int N, int K,
                  int relu) {
    dim3 grid(kBT / 128, (N + 63) / 64);
    gemm_kernel<<<grid, 256, 0, stream>>>(A, Wt, bias, resid, oF, oB, N, K,
                                          relu);
  };
  auto ln = [&](const float* xin, const float* g, const float* b,
                bf16_t* outb) {
    ln_kernel<<<kBT / 8, 256, 0, stream>>>(xin, g, b, outb, kBT);
  };

  for (int l = 0; l < kL; ++l) {
    size_t oEE = (size_t)l * kE * kE;
    size_t oEF = (size_t)l * kE * kF;
    const float* bq_l = bq + (size_t)l * kE;
    const float* bk_l = bk + (size_t)l * kE;
    const float* bv_l = bv + (size_t)l * kE;
    const float* bo_l = bo + (size_t)l * kE;
    const float* c1_l = c1 + (size_t)l * kF;
    const float* c2_l = c2 + (size_t)l * kE;

    ln(x, g1 + (size_t)l * kE, be1 + (size_t)l * kE, hbf);
    gemm(hbf, Wqt + oEE, bq_l, nullptr, nullptr, qb, kE, kE, 0);
    gemm(hbf, Wkt + oEE, bk_l, nullptr, nullptr, kb, kE, kE, 0);
    gemm(hbf, Wvt + oEE, bv_l, nullptr, nullptr, vb, kE, kE, 0);

    attn_kernel<<<dim3(kB * kH, kT / 64), 256, 0, stream>>>(qb, kb, vb, ob);

    gemm(ob, Wot + oEE, bo_l, x, x, nullptr, kE, kE, 0);

    ln(x, g2 + (size_t)l * kE, be2 + (size_t)l * kE, hbf);
    gemm(hbf, W1t + oEF, c1_l, nullptr, nullptr, midb, kF, kE, 1);
    gemm(midb, W2t + oEF, c2_l, x, x, nullptr, kE, kF, 0);
  }

  {
    size_t tot = (size_t)kBT * kE;
    f2b_kernel<<<(int)((tot + 255) / 256), 256, 0, stream>>>(x, hbf, tot);
  }
  gemm(hbf, Wlt, bl, nullptr, out, nullptr, kV, kE, 0);
}
